// DeepKalmanFilter_38654705664177
// MI455X (gfx1250) — compile-verified, matrix operand order checked by reading, not run on hardware
//
#include <hip/hip_runtime.h>

// ---------- types ----------
typedef __attribute__((ext_vector_type(16))) __bf16    v16bf;
typedef __attribute__((ext_vector_type(8)))  float     v8f;
typedef __attribute__((ext_vector_type(4)))  unsigned  v4u;

#define WMMA_BF16(a, b, c) \
  __builtin_amdgcn_wmma_f32_16x16x32_bf16(false, (a), false, (b), (short)0, (c), false, false)

// Problem geometry (compile-time; powers of two -> all strides fold to shifts/imm offsets).
#define GB   32          // batch
#define GT   1024        // timesteps
#define GH   1024        // hidden
#define GG   (4 * GH)    // 4H = gate width
#define LOGB 5
#define LOGT 10

// Load a 16x32 bf16 fragment (A-layout / B^T-layout) from row-major memory.
template <int LD>
__device__ __forceinline__ v16bf load_frag_g(const __bf16* __restrict__ base) {
  const int lane = threadIdx.x & 31;
  const int r  = lane & 15;
  const int hl = lane >> 4;
  const __bf16* p = base + r * LD + hl * 8;
  union { v4u u[2]; v16bf f; } u;
  u.u[0] = *(const v4u*)(p);        // K = hl*8 .. hl*8+7
  u.u[1] = *(const v4u*)(p + 16);   // K = 16+hl*8 ..
  return u.f;
}

__device__ __forceinline__ v16bf load_frag_lds(const char* ldsB, int nf, int ks) {
  const int lane = threadIdx.x & 31;
  const v4u* p = (const v4u*)(ldsB + ((size_t)((nf * 32 + ks) * 32 + lane) << 5));
  union { v4u u[2]; v16bf f; } u;
  u.u[0] = p[0];
  u.u[1] = p[1];
  return u.f;
}

// ---------- small utility kernels ----------
__global__ void zero_u32_kernel(unsigned* p, int n) {
  int i = blockIdx.x * blockDim.x + threadIdx.x;
  if (i < n) p[i] = 0u;
}

// obs [B,T,D] f32 -> obsb [T,B,D] bf16 (time-major, coalesced writes)
__global__ void obs_to_tmajor_bf16_kernel(const float* __restrict__ s, __bf16* __restrict__ d,
                                          int D, int total) {
  int i = blockIdx.x * blockDim.x + threadIdx.x;
  if (i < total) {
    int dcol = i % D;
    int r    = i / D;              // r = t*B + b (output row)
    int t    = r >> LOGB;
    int b    = r & (GB - 1);
    d[i] = (__bf16)s[(size_t)((b << LOGT) | t) * D + dcol];
  }
}

// dst[C x R] (bf16) = transpose(src[R x C] f32)
__global__ void transpose_bf16_kernel(const float* __restrict__ src, __bf16* __restrict__ dst,
                                      int R, int C) {
  int i = blockIdx.x * blockDim.x + threadIdx.x;
  if (i < R * C) {
    int r = i / C, c = i - r * C;
    dst[(size_t)c * R + r] = (__bf16)src[i];
  }
}

// ---------- GEMM: C[M,N] = A[M,K] * Bt[N,K]^T + bias ----------
// EP:   0 = f32 out, 1 = f32 out + softplus, 2 = bf16 out
// PERM: 0 = identity rows; 2 = rows are (t*B+b), store to (b*T+t)  [heads]
// K, N compile-time so fragment addresses fold into shifts + immediate offsets.
// block = 256 threads = 8 waves (4 M-waves x 2 N-waves); block tile 128x128;
// wave tile 32x64 = 2 m-frags x 4 n-frags.
template <int EP, int PERM, int K, int N>
__global__ void gemm_bf16_kernel(const __bf16* __restrict__ A, const __bf16* __restrict__ Bt,
                                 const float* __restrict__ bias, void* __restrict__ Cout) {
  const int wave = threadIdx.x >> 5;
  const int lane = threadIdx.x & 31;
  const int wm = wave & 3, wn = wave >> 2;
  const int mbase = blockIdx.y * 128 + wm * 32;
  const int nbase = blockIdx.x * 128 + wn * 64;

  const __bf16* Arow0 = A + (size_t)mbase * K;
  const __bf16* Arow1 = A + (size_t)(mbase + 16) * K;
  const __bf16* Brow  = Bt + (size_t)nbase * K;

  v8f acc[2][4] = {};
#pragma unroll 2
  for (int k = 0; k < K; k += 32) {
    if (k + 128 < K) {
      __builtin_prefetch(Arow0 + k + 128, 0, 1);
      __builtin_prefetch(Brow + k + 128, 0, 1);
    }
    v16bf a0 = load_frag_g<K>(Arow0 + k);
    v16bf a1 = load_frag_g<K>(Arow1 + k);
#pragma unroll
    for (int nf = 0; nf < 4; ++nf) {
      v16bf b = load_frag_g<K>(Brow + nf * 16 * K + k);
      acc[0][nf] = WMMA_BF16(a0, b, acc[0][nf]);
      acc[1][nf] = WMMA_BF16(a1, b, acc[1][nf]);
    }
  }

  const int n0 = lane & 15, hl = lane >> 4;
#pragma unroll
  for (int mf = 0; mf < 2; ++mf)
#pragma unroll
    for (int nf = 0; nf < 4; ++nf) {
      const int col = nbase + nf * 16 + n0;
      const float bcol = bias[col];
#pragma unroll
      for (int r = 0; r < 8; ++r) {
        const int row = mbase + mf * 16 + hl * 8 + r;
        int orow;
        if (PERM == 2) {           // row = t*B+b -> orow = b*T+t
          orow = ((row & (GB - 1)) << LOGT) | (row >> LOGB);
        } else {
          orow = row;
        }
        float v = acc[mf][nf][r] + bcol;
        if (EP == 1) v = (v > 20.f) ? v : log1pf(__expf(v));
        if (EP == 2) ((__bf16*)Cout)[(size_t)orow * N + col] = (__bf16)v;
        else         ((float*)Cout)[(size_t)orow * N + col] = v;
      }
    }
}

// ---------- persistent LSTM recurrence (time-major buffers) ----------
// xz   : [T, B, 4H] bf16   hseq : [T, B, H] bf16   (all geometry compile-time)
// grid = H/32 = 32 workgroups; workgroup j owns h-columns [j*32, j*32+32).
// LDS: Wh slice in fragment order (256KB) + zbuf 32x128 f32 (16KB) + c 32x32 f32 (4KB).
__global__ void lstm_rec_kernel(const __bf16* __restrict__ xz,
                                const __bf16* __restrict__ WhT /*[4H, H]*/,
                                __bf16* __restrict__ hseq,
                                unsigned* __restrict__ bar) {
  extern __shared__ char smem[];
  char*  ldsB = smem;                               // 256 KB
  float* zbuf = (float*)(smem + 256 * 1024);        // 16 KB : 32 rows x 128 cols
  float* cbuf = (float*)(smem + 272 * 1024);        // 4 KB  : 32 rows x 32 cols

  const int tid  = threadIdx.x;
  const int lane = tid & 31;
  const int wave = tid >> 5;
  const int wg   = blockIdx.x;          // h-column tile

  // Stage this WG's Wh slice into LDS in fragment order. Wave w fills nf == w.
  {
    const int nf = wave;
    const int g = nf >> 1, u = nf & 1;
    const int zcol = g * GH + wg * 32 + u * 16;     // base z column of this frag
    const int nn = lane & 15, hh = lane >> 4;
    const __bf16* src = WhT + (size_t)(zcol + nn) * GH + hh * 8;
    for (int ks = 0; ks < 32; ++ks) {
      v4u lo = *(const v4u*)(src + ks * 32);
      v4u hi = *(const v4u*)(src + ks * 32 + 16);
      v4u* dst = (v4u*)(ldsB + ((size_t)((nf * 32 + ks) * 32 + lane) << 5));
      dst[0] = lo;
      dst[1] = hi;
    }
  }
  for (int i = tid; i < 32 * 32; i += 256) cbuf[i] = 0.f;   // c0 = 0
  __syncthreads();

  const int m   = wave >> 2;            // m-frag (batch rows 16m..16m+15)
  const int nf0 = (wave & 3) * 2;       // first of two n-frags
  const int n = lane & 15, hl = lane >> 4;

  for (int t = 0; t < GT; ++t) {
    v8f acc0 = {}, acc1 = {};
    if (t > 0) {
      // A tile: rows (t-1)*B + (m*16 .. m*16+15), contiguous block, ld = H
      const __bf16* abase = hseq + ((size_t)(t - 1) * GB + m * 16) * GH;
#pragma unroll 4
      for (int ks = 0; ks < 32; ++ks) {
        v16bf a  = load_frag_g<GH>(abase + ks * 32);
        v16bf b0 = load_frag_lds(ldsB, nf0, ks);
        v16bf b1 = load_frag_lds(ldsB, nf0 + 1, ks);
        acc0 = WMMA_BF16(a, b0, acc0);
        acc1 = WMMA_BF16(a, b1, acc1);
      }
    }
    // z = acc + xz_t  -> zbuf (local cols: gate g occupies [g*32, g*32+32))
    {
      v8f accs[2] = {acc0, acc1};
#pragma unroll
      for (int f = 0; f < 2; ++f) {
        const int nf = nf0 + f;
        const int g = nf >> 1, u = nf & 1;
        const int zc = g * GH + wg * 32 + u * 16 + n;
        // rows t*B + m*16 + hl*8 + r; single base, immediate offsets r*GG
        const __bf16* xzb = xz + ((size_t)t * GB + m * 16 + hl * 8) * GG + zc;
#pragma unroll
        for (int r = 0; r < 8; ++r) {
          const int b = m * 16 + hl * 8 + r;
          float z = accs[f][r] + (float)xzb[r * GG];
          zbuf[b * 128 + nf * 16 + n] = z;
        }
      }
    }
    __syncthreads();
    // fused gates: 32 rows x 32 h-cols, 4 per thread
    for (int i = tid; i < 32 * 32; i += 256) {
      const int b = i >> 5, c = i & 31;
      const float zi = zbuf[b * 128 + c];
      const float zf = zbuf[b * 128 + 32 + c];
      const float zg = zbuf[b * 128 + 64 + c];
      const float zo = zbuf[b * 128 + 96 + c];
      const float ig = 1.f / (1.f + __expf(-zi));
      const float fg = 1.f / (1.f + __expf(-zf));
      const float gg = tanhf(zg);
      const float og = 1.f / (1.f + __expf(-zo));
      const float cn = fg * cbuf[i] + ig * gg;
      cbuf[i] = cn;
      hseq[((size_t)t * GB + b) * GH + wg * 32 + c] = (__bf16)(og * tanhf(cn));
    }
    __syncthreads();
    // grid-wide step barrier (32 WGs, all resident)
    if (tid == 0) {
      __threadfence();
      atomicAdd(bar, 1u);
      const unsigned goal = (unsigned)(t + 1) * gridDim.x;
      while (*((volatile unsigned*)bar) < goal) { __builtin_amdgcn_s_sleep(2); }
      __threadfence();
    }
    __syncthreads();
  }
}

// ---------- host launch ----------
extern "C" void kernel_launch(void* const* d_in, const int* in_sizes, int n_in,
                              void* d_out, int out_size, void* d_ws, size_t ws_size,
                              hipStream_t stream) {
  const float* obs = (const float*)d_in[0];
  const float* W1x = (const float*)d_in[1];
  const float* W1h = (const float*)d_in[2];
  const float* b1  = (const float*)d_in[3];
  const float* W2x = (const float*)d_in[4];
  const float* W2h = (const float*)d_in[5];
  const float* b2  = (const float*)d_in[6];
  const float* Wm  = (const float*)d_in[7];
  const float* bm  = (const float*)d_in[8];
  const float* Wc  = (const float*)d_in[9];
  const float* bc  = (const float*)d_in[10];

  constexpr int B = GB, T = GT, DOBS = 256, H = GH, DS = 512;
  constexpr int G = GG;                     // 4096
  constexpr size_t M = (size_t)B * T;       // 32768

  char* ws = (char*)d_ws;
  auto alloc = [&](size_t bytes) -> char* {
    char* p = ws;
    ws += (bytes + 255) & ~(size_t)255;
    return p;
  };
  __bf16* obsb = (__bf16*)alloc(M * DOBS * 2);        // [T,B,DOBS]
  __bf16* W1xT = (__bf16*)alloc((size_t)G * DOBS * 2);
  __bf16* W1hT = (__bf16*)alloc((size_t)G * H * 2);
  __bf16* W2xT = (__bf16*)alloc((size_t)G * H * 2);
  __bf16* W2hT = (__bf16*)alloc((size_t)G * H * 2);
  __bf16* WmT  = (__bf16*)alloc((size_t)DS * H * 2);
  __bf16* WcT  = (__bf16*)alloc((size_t)DS * H * 2);
  __bf16* xz   = (__bf16*)alloc(M * G * 2);           // [T,B,4H], shared by layers
  __bf16* h1   = (__bf16*)alloc(M * H * 2);           // [T,B,H]
  __bf16* h2   = (__bf16*)alloc(M * H * 2);           // [T,B,H]
  unsigned* bar = (unsigned*)alloc(512);

  const int TPB = 256;
  zero_u32_kernel<<<1, 128, 0, stream>>>(bar, 128);

  // weight transposes -> bf16 [N x K]
  transpose_bf16_kernel<<<(DOBS * G + TPB - 1) / TPB, TPB, 0, stream>>>(W1x, W1xT, DOBS, G);
  transpose_bf16_kernel<<<(H * G + TPB - 1) / TPB, TPB, 0, stream>>>(W1h, W1hT, H, G);
  transpose_bf16_kernel<<<(H * G + TPB - 1) / TPB, TPB, 0, stream>>>(W2x, W2xT, H, G);
  transpose_bf16_kernel<<<(H * G + TPB - 1) / TPB, TPB, 0, stream>>>(W2h, W2hT, H, G);
  transpose_bf16_kernel<<<(H * DS + TPB - 1) / TPB, TPB, 0, stream>>>(Wm, WmT, H, DS);
  transpose_bf16_kernel<<<(H * DS + TPB - 1) / TPB, TPB, 0, stream>>>(Wc, WcT, H, DS);
  obs_to_tmajor_bf16_kernel<<<(int)((M * DOBS + TPB - 1) / TPB), TPB, 0, stream>>>(
      obs, obsb, DOBS, (int)(M * DOBS));

  const size_t SMEM = 276 * 1024;

  // Layer 1: xz = obs_tm @ W1x + b1 (identity rows: t*B+b) ; recurrence
  gemm_bf16_kernel<2, 0, DOBS, G><<<dim3(G / 128, (int)(M / 128)), TPB, 0, stream>>>(obsb, W1xT, b1, xz);
  lstm_rec_kernel<<<H / 32, TPB, SMEM, stream>>>(xz, W1hT, h1, bar);

  // Layer 2: xz = h1 @ W2x + b2 ; recurrence
  gemm_bf16_kernel<2, 0, H, G><<<dim3(G / 128, (int)(M / 128)), TPB, 0, stream>>>(h1, W2xT, b2, xz);
  lstm_rec_kernel<<<H / 32, TPB, SMEM, stream>>>(xz, W2hT, h2, bar + 64);

  // Heads: time-major rows -> [B,T,DS] output via permuted store
  float* meanp = (float*)d_out;
  float* covp  = meanp + M * DS;
  gemm_bf16_kernel<0, 2, H, DS><<<dim3(DS / 128, (int)(M / 128)), TPB, 0, stream>>>(h2, WmT, bm, meanp);
  gemm_bf16_kernel<1, 2, H, DS><<<dim3(DS / 128, (int)(M / 128)), TPB, 0, stream>>>(h2, WcT, bc, covp);
}
